// PointTransformerCls_6442450944540
// MI455X (gfx1250) — matrix operand order repaired and checked by reading, not compiled, not run on hardware
//
#include <hip/hip_runtime.h>

typedef unsigned short u16;
typedef unsigned int   u32;
typedef __attribute__((ext_vector_type(16))) __bf16 bf16x16;
typedef __attribute__((ext_vector_type(8)))  float  f32x8;

union FragB {
    bf16x16 v;
    uint4   q[2];
};

#define F8Z {0.f, 0.f, 0.f, 0.f, 0.f, 0.f, 0.f, 0.f}

__device__ __forceinline__ u16 f2bf(float x) {
    u32 u = __float_as_uint(x);
    u32 r = (u + 0x7FFFu + ((u >> 16) & 1u)) >> 16;
    return (u16)r;
}
__device__ __forceinline__ float bf2f(u16 h) {
    return __uint_as_float(((u32)h) << 16);
}
__device__ __forceinline__ f32x8 wmma_bf16(bf16x16 a, bf16x16 b, f32x8 c) {
    // 8 args: (neg_a, A, neg_b, B, c_mod, C, reuse_a, reuse_b)
    return __builtin_amdgcn_wmma_f32_16x16x32_bf16(false, a, false, b, (short)0, c, false, false);
}

// gfx1250 async global->LDS copy path (ASYNCcnt), with synchronous fallback.
#if defined(__HIP_DEVICE_COMPILE__)
#  if __has_builtin(__builtin_amdgcn_global_load_async_to_lds_b128) && \
      __has_builtin(__builtin_amdgcn_s_wait_asynccnt)
#    define HAVE_ASYNC 1
#  endif
#endif
#ifndef HAVE_ASYNC
#  define HAVE_ASYNC 0
#endif

#if HAVE_ASYNC
typedef int v4i __attribute__((vector_size(4 * sizeof(int))));
typedef __attribute__((address_space(1))) v4i* gas_v4i_ptr;  // global (AS1)
typedef __attribute__((address_space(3))) v4i* las_v4i_ptr;  // LDS (AS3)
#endif

__device__ __forceinline__ void async_copy16(u16* lds_dst, const u16* gsrc) {
#if HAVE_ASYNC
    __builtin_amdgcn_global_load_async_to_lds_b128(
        (gas_v4i_ptr)(gsrc), (las_v4i_ptr)(lds_dst), 0, 0);
#else
    *(uint4*)lds_dst = *(const uint4*)gsrc;
#endif
}
__device__ __forceinline__ void async_wait_all() {
#if HAVE_ASYNC
    __builtin_amdgcn_s_wait_asynccnt(0);
#endif
}

#define BN_RS 0.99999500003749972f /* 1/sqrt(1+1e-5) */

// ---------------------------------------------------------------------------
// Weight transpose + f32 -> bf16 convert: W[K][N] -> Wt[N][Kp] (zero-pad K..Kp)
// ---------------------------------------------------------------------------
__global__ void tr_kernel(const float* __restrict__ W, u16* __restrict__ Wt,
                          int K, int N, int Kp) {
    int t = blockIdx.x * 256 + threadIdx.x;
    if (t >= N * Kp) return;
    int n = t / Kp, k = t - n * Kp;
    float v = (k < K) ? W[(size_t)k * N + n] : 0.f;
    Wt[t] = f2bf(v);
}

// ---------------------------------------------------------------------------
// x[B,N,6] -> xyz[P][3], featb[P][512] bf16 (zero-padded beyond 6)
// ---------------------------------------------------------------------------
__global__ void init_kernel(const float* __restrict__ x, float* __restrict__ xyz,
                            u16* __restrict__ featb) {
    int p = blockIdx.x;
    int t = threadIdx.x;           // 512 threads
    if (t < 3) xyz[p * 3 + t] = x[p * 6 + t];
    featb[(size_t)p * 512 + t] = (t < 6) ? f2bf(x[p * 6 + t]) : (u16)0;
}

// ---------------------------------------------------------------------------
// Brute-force KNN, one query point per thread, batch xyz staged in LDS.
// ---------------------------------------------------------------------------
template <int KK>
__global__ __launch_bounds__(256) void knn_kernel(const float* __restrict__ xyz,
                                                  int* __restrict__ idx) {
    __shared__ float sp[3 * 1024];
    int b = blockIdx.x >> 2;
    int n = ((blockIdx.x & 3) << 8) + threadIdx.x;
    for (int j = threadIdx.x; j < 1024; j += 256) {
        sp[j]        = xyz[(b * 1024 + j) * 3 + 0];
        sp[1024 + j] = xyz[(b * 1024 + j) * 3 + 1];
        sp[2048 + j] = xyz[(b * 1024 + j) * 3 + 2];
    }
    __syncthreads();
    float qx = sp[n], qy = sp[1024 + n], qz = sp[2048 + n];
    float bd[KK];
    int   bi[KK];
#pragma unroll
    for (int t = 0; t < KK; ++t) { bd[t] = 3.4e38f; bi[t] = 0; }
    for (int j = 0; j < 1024; ++j) {
        float dx = qx - sp[j], dy = qy - sp[1024 + j], dz = qz - sp[2048 + j];
        float d = dx * dx + dy * dy + dz * dz;
        if (d < bd[KK - 1]) {
            int t = KK - 1;
            while (t > 0 && bd[t - 1] > d) { bd[t] = bd[t - 1]; bi[t] = bi[t - 1]; --t; }
            bd[t] = d; bi[t] = j;
        }
    }
    int* op = idx + (size_t)(b * 1024 + n) * 16;
#pragma unroll
    for (int t = 0; t < KK; ++t) op[t] = bi[t];
#pragma unroll
    for (int t = KK; t < 16; ++t) op[t] = bi[0];
}

// ---------------------------------------------------------------------------
// Transition down (fused): A = gathered 16 x Kp neighbor rows, WMMA pairs of
// column tiles (A reused), BN+ReLU, masked max over rows.
// ---------------------------------------------------------------------------
__global__ __launch_bounds__(256) void td_kernel(
    const u16* __restrict__ featb, const int* __restrict__ idx,
    const u16* __restrict__ Wt, const float* __restrict__ g, const float* __restrict__ bv,
    float* __restrict__ outf, u16* __restrict__ outb, int c, int Kp, int ns) {
    int pt = blockIdx.x;
    int b = pt >> 10;
    int lane = threadIdx.x & 31, wave = threadIdx.x >> 5;
    int l = lane & 15, half = lane >> 4;
    int nbr = idx[(size_t)pt * 16 + l];
    const u16* arow = featb + (size_t)(b * 1024 + nbr) * 512;
    int npairs = c >> 5;
    for (int p = wave; p < npairs; p += 8) {
        int t0 = 2 * p;
        f32x8 acc0 = F8Z, acc1 = F8Z;
        const u16* br0 = Wt + (size_t)(t0 * 16 + l) * Kp;
        const u16* br1 = br0 + (size_t)16 * Kp;
        for (int k0 = 0; k0 < Kp; k0 += 32) {
            FragB a, b0, b1;
            a.q[0]  = *(const uint4*)(arow + k0 + 16 * half);
            a.q[1]  = *(const uint4*)(arow + k0 + 16 * half + 8);
            b0.q[0] = *(const uint4*)(br0 + k0 + 16 * half);
            b0.q[1] = *(const uint4*)(br0 + k0 + 16 * half + 8);
            b1.q[0] = *(const uint4*)(br1 + k0 + 16 * half);
            b1.q[1] = *(const uint4*)(br1 + k0 + 16 * half + 8);
            acc0 = wmma_bf16(a.v, b0.v, acc0);
            acc1 = wmma_bf16(a.v, b1.v, acc1);
        }
        auto reduce_store = [&](const f32x8& acc, int col) {
            float sc = g[col] * BN_RS, bi = bv[col];
            float mx = -1e30f;
#pragma unroll
            for (int r = 0; r < 8; ++r) {
                int m = r + 8 * half;
                float v = fmaxf(acc[r] * sc + bi, 0.f);
                if (m < ns) mx = fmaxf(mx, v);
            }
            mx = fmaxf(mx, __shfl_xor(mx, 16, 32));
            if (half == 0) {
                outf[(size_t)pt * 512 + col] = mx;
                outb[(size_t)pt * 512 + col] = f2bf(mx);
            }
        };
        reduce_store(acc0, t0 * 16 + l);
        reduce_store(acc1, t0 * 16 + 16 + l);
    }
}

// ---------------------------------------------------------------------------
// Generic WMMA GEMM, 32x32 register tile per wave (2 A frags x 2 B frags ->
// 4 WMMAs per K step).  grid = (N/32, M/256), block = 256 (8 waves).
// ---------------------------------------------------------------------------
__device__ __forceinline__ void gemm_epi(const f32x8& acc, int m0, int half, int col,
                                         const float* bias, const float* res, int ldr,
                                         float* C, int ldc, u16* Cb, int ldcb, int relu) {
    float bvv = bias ? bias[col] : 0.f;
#pragma unroll
    for (int r = 0; r < 8; ++r) {
        int m = m0 + r + 8 * half;
        float v = acc[r] + bvv;
        if (res) v += res[(size_t)m * ldr + col];
        if (relu) v = fmaxf(v, 0.f);
        if (C) C[(size_t)m * ldc + col] = v;
        if (Cb) Cb[(size_t)m * ldcb + col] = f2bf(v);
    }
}

__global__ __launch_bounds__(256) void gemm_kernel(
    const u16* __restrict__ A, int lda,
    const u16* __restrict__ W, int ldw,
    const float* __restrict__ bias,
    const float* __restrict__ res, int ldr,
    float* __restrict__ C, int ldc,
    u16* __restrict__ Cb, int ldcb,
    int M, int K, int relu) {
    int wave = threadIdx.x >> 5;
    int lane = threadIdx.x & 31;
    int l = lane & 15, half = lane >> 4;
    int n0 = blockIdx.x * 32;
    int m0 = (blockIdx.y * 8 + wave) * 32;
    if (m0 >= M) return;
    const u16* arow0 = A + (size_t)(m0 + l) * lda;
    const u16* arow1 = A + (size_t)(m0 + 16 + l) * lda;
    const u16* brow0 = W + (size_t)(n0 + l) * ldw;
    const u16* brow1 = W + (size_t)(n0 + 16 + l) * ldw;
    f32x8 acc00 = F8Z, acc01 = F8Z, acc10 = F8Z, acc11 = F8Z;
    for (int k0 = 0; k0 < K; k0 += 32) {
        FragB a0, a1, b0, b1;
        a0.q[0] = *(const uint4*)(arow0 + k0 + 16 * half);
        a0.q[1] = *(const uint4*)(arow0 + k0 + 16 * half + 8);
        a1.q[0] = *(const uint4*)(arow1 + k0 + 16 * half);
        a1.q[1] = *(const uint4*)(arow1 + k0 + 16 * half + 8);
        b0.q[0] = *(const uint4*)(brow0 + k0 + 16 * half);
        b0.q[1] = *(const uint4*)(brow0 + k0 + 16 * half + 8);
        b1.q[0] = *(const uint4*)(brow1 + k0 + 16 * half);
        b1.q[1] = *(const uint4*)(brow1 + k0 + 16 * half + 8);
        acc00 = wmma_bf16(a0.v, b0.v, acc00);
        acc01 = wmma_bf16(a0.v, b1.v, acc01);
        acc10 = wmma_bf16(a1.v, b0.v, acc10);
        acc11 = wmma_bf16(a1.v, b1.v, acc11);
    }
    gemm_epi(acc00, m0,      half, n0 + l,      bias, res, ldr, C, ldc, Cb, ldcb, relu);
    gemm_epi(acc01, m0,      half, n0 + 16 + l, bias, res, ldr, C, ldc, Cb, ldcb, relu);
    gemm_epi(acc10, m0 + 16, half, n0 + l,      bias, res, ldr, C, ldc, Cb, ldcb, relu);
    gemm_epi(acc11, m0 + 16, half, n0 + 16 + l, bias, res, ldr, C, ldc, Cb, ldcb, relu);
}

// ---------------------------------------------------------------------------
// Fused per-point attention block. grid = P blocks, 256 threads, dyn LDS.
// kf/vf gathers staged to LDS via gfx1250 async global->LDS (ASYNCcnt),
// overlapped with the pos-MLP / a2 WMMA loops. LDS regions are aliased:
// s_kf lives in the not-yet-written h region, s_vf in the dead a region.
// ---------------------------------------------------------------------------
__global__ __launch_bounds__(256) void attn_kernel(
    const int* __restrict__ idx, const float* __restrict__ xyz,
    const float* __restrict__ q, const u16* __restrict__ kf, const u16* __restrict__ vf,
    const float* __restrict__ pW1, const float* __restrict__ pb1,
    const u16* __restrict__ pW2t, const float* __restrict__ pb2,
    const u16* __restrict__ aW1t, const float* __restrict__ ab1,
    const u16* __restrict__ aW2t, const float* __restrict__ ab2,
    float* __restrict__ nf, u16* __restrict__ nfb,
    int c, int ns, int cshift /* log2(c/8) */) {
    extern __shared__ char smem[];
    int*   s_idx = (int*)smem;                           // 16 ints
    float* s_rel = (float*)(smem + 64);                  // 16 x 4
    u16*   s_p1  = (u16*)(smem + 64 + 256);              // 16 x 64 bf16
    size_t off = 64 + 256 + 2048;
    float* s_pos = (float*)(smem + off); off += (size_t)16 * c * 4;
    u16*   s_a   = (u16*)(smem + off);   off += (size_t)16 * c * 2;
    u16*   s_h   = (u16*)(smem + off);   off += (size_t)16 * 4 * c * 2;
    float* s_a2  = (float*)(smem + off);
    u16*   s_kf  = s_h;   // alias: h not written until after a is built
    u16*   s_vf  = s_a;   // alias: a dead once h is computed

    int pt = blockIdx.x;
    int bb = pt >> 10;
    int tid = threadIdx.x;
    int lane = tid & 31, wave = tid >> 5;
    int l = lane & 15, half = lane >> 4;
    int c4 = 4 * c;
    int chunks = 16 << cshift;     // 16 rows * (c/8) 16-byte chunks

    if (tid < 16) {
        int nb = idx[(size_t)pt * 16 + tid];
        s_idx[tid] = nb;
        s_rel[tid * 4 + 0] = xyz[pt * 3 + 0] - xyz[(bb * 1024 + nb) * 3 + 0];
        s_rel[tid * 4 + 1] = xyz[pt * 3 + 1] - xyz[(bb * 1024 + nb) * 3 + 1];
        s_rel[tid * 4 + 2] = xyz[pt * 3 + 2] - xyz[(bb * 1024 + nb) * 3 + 2];
        s_rel[tid * 4 + 3] = 0.f;
    }
    __syncthreads();

    // issue async staging of gathered kf rows (completion awaited before a-build)
    for (int e = tid; e < chunks; e += 256) {
        int r = e >> cshift;
        int ch0 = (e - (r << cshift)) << 3;
        async_copy16(s_kf + r * c + ch0,
                     kf + (size_t)(bb * 1024 + s_idx[r]) * 512 + ch0);
    }

    // p1 = relu(rel @ pW1 + pb1): 16x64 (K=3, VALU)
    for (int e = tid; e < 16 * 64; e += 256) {
        int r = e >> 6, j = e & 63;
        float s = pb1[j];
#pragma unroll
        for (int d = 0; d < 3; ++d) s += s_rel[r * 4 + d] * pW1[d * 64 + j];
        s_p1[e] = f2bf(fmaxf(s, 0.f));
    }
    __syncthreads();

    // pos = p1 @ pW2t^T + pb2: 16 x c, K=64 (paired tiles share A frags)
    int npc = c >> 5;              // pairs of 16-col tiles
    for (int p = wave; p < npc; p += 8) {
        int t0 = 2 * p;
        f32x8 acc0 = F8Z, acc1 = F8Z;
        const u16* ar  = s_p1 + l * 64;
        const u16* br0 = pW2t + (size_t)(t0 * 16 + l) * 64;
        const u16* br1 = br0 + (size_t)16 * 64;
        for (int k0 = 0; k0 < 64; k0 += 32) {
            FragB a, b0, b1;
            a.q[0]  = *(const uint4*)(ar + k0 + 16 * half);
            a.q[1]  = *(const uint4*)(ar + k0 + 16 * half + 8);
            b0.q[0] = *(const uint4*)(br0 + k0 + 16 * half);
            b0.q[1] = *(const uint4*)(br0 + k0 + 16 * half + 8);
            b1.q[0] = *(const uint4*)(br1 + k0 + 16 * half);
            b1.q[1] = *(const uint4*)(br1 + k0 + 16 * half + 8);
            acc0 = wmma_bf16(a.v, b0.v, acc0);
            acc1 = wmma_bf16(a.v, b1.v, acc1);
        }
#pragma unroll
        for (int r = 0; r < 8; ++r) {
            int m = r + 8 * half;
            s_pos[m * c + t0 * 16 + l]      = acc0[r] + pb2[t0 * 16 + l];
            s_pos[m * c + t0 * 16 + 16 + l] = acc1[r] + pb2[t0 * 16 + 16 + l];
        }
    }
    __syncthreads();
    async_wait_all();              // kf rows now resident in LDS
    __syncthreads();

    // a = q - kf + pos (bf16 in LDS), invalid rows zeroed
    for (int e = tid; e < 16 * c; e += 256) {
        int r = e / c, ch = e - r * c;
        float v = 0.f;
        if (r < ns) {
            float qv = q[(size_t)pt * 512 + ch];
            v = qv - bf2f(s_kf[e]) + s_pos[e];
        }
        s_a[e] = f2bf(v);
    }
    __syncthreads();

    // h = relu(a @ aW1t^T + ab1): 16 x 4c, K=c (paired tiles, overwrites s_kf)
    int np4 = c4 >> 5;
    for (int p = wave; p < np4; p += 8) {
        int t0 = 2 * p;
        f32x8 acc0 = F8Z, acc1 = F8Z;
        const u16* ar  = s_a + (size_t)l * c;
        const u16* br0 = aW1t + (size_t)(t0 * 16 + l) * c;
        const u16* br1 = br0 + (size_t)16 * c;
        for (int k0 = 0; k0 < c; k0 += 32) {
            FragB a, b0, b1;
            a.q[0]  = *(const uint4*)(ar + k0 + 16 * half);
            a.q[1]  = *(const uint4*)(ar + k0 + 16 * half + 8);
            b0.q[0] = *(const uint4*)(br0 + k0 + 16 * half);
            b0.q[1] = *(const uint4*)(br0 + k0 + 16 * half + 8);
            b1.q[0] = *(const uint4*)(br1 + k0 + 16 * half);
            b1.q[1] = *(const uint4*)(br1 + k0 + 16 * half + 8);
            acc0 = wmma_bf16(a.v, b0.v, acc0);
            acc1 = wmma_bf16(a.v, b1.v, acc1);
        }
#pragma unroll
        for (int r = 0; r < 8; ++r) {
            int m = r + 8 * half;
            s_h[(size_t)m * c4 + t0 * 16 + l] =
                f2bf(fmaxf(acc0[r] + ab1[t0 * 16 + l], 0.f));
            s_h[(size_t)m * c4 + t0 * 16 + 16 + l] =
                f2bf(fmaxf(acc1[r] + ab1[t0 * 16 + 16 + l], 0.f));
        }
    }
    __syncthreads();

    // issue async staging of gathered vf rows into dead s_a region;
    // overlaps with the a2 WMMA loop below.
    for (int e = tid; e < chunks; e += 256) {
        int r = e >> cshift;
        int ch0 = (e - (r << cshift)) << 3;
        async_copy16(s_vf + r * c + ch0,
                     vf + (size_t)(bb * 1024 + s_idx[r]) * 512 + ch0);
    }

    // a2 = (h @ aW2t^T + ab2) / sqrt(c): 16 x c, K=4c (paired tiles)
    float scale = rsqrtf((float)c);
    for (int p = wave; p < npc; p += 8) {
        int t0 = 2 * p;
        f32x8 acc0 = F8Z, acc1 = F8Z;
        const u16* ar  = s_h + (size_t)l * c4;
        const u16* br0 = aW2t + (size_t)(t0 * 16 + l) * c4;
        const u16* br1 = br0 + (size_t)16 * c4;
        for (int k0 = 0; k0 < c4; k0 += 32) {
            FragB a, b0, b1;
            a.q[0]  = *(const uint4*)(ar + k0 + 16 * half);
            a.q[1]  = *(const uint4*)(ar + k0 + 16 * half + 8);
            b0.q[0] = *(const uint4*)(br0 + k0 + 16 * half);
            b0.q[1] = *(const uint4*)(br0 + k0 + 16 * half + 8);
            b1.q[0] = *(const uint4*)(br1 + k0 + 16 * half);
            b1.q[1] = *(const uint4*)(br1 + k0 + 16 * half + 8);
            acc0 = wmma_bf16(a.v, b0.v, acc0);
            acc1 = wmma_bf16(a.v, b1.v, acc1);
        }
#pragma unroll
        for (int r = 0; r < 8; ++r) {
            int m = r + 8 * half;
            s_a2[m * c + t0 * 16 + l]      = (acc0[r] + ab2[t0 * 16 + l]) * scale;
            s_a2[m * c + t0 * 16 + 16 + l] = (acc1[r] + ab2[t0 * 16 + 16 + l]) * scale;
        }
    }
    async_wait_all();              // vf rows resident
    __syncthreads();

    // softmax over rows r<ns per column, then sum attn * (vf + pos)
    for (int ch = tid; ch < c; ch += 256) {
        float mx = -1e30f;
        for (int r = 0; r < ns; ++r) mx = fmaxf(mx, s_a2[r * c + ch]);
        float sum = 0.f;
        for (int r = 0; r < ns; ++r) sum += __expf(s_a2[r * c + ch] - mx);
        float inv = 1.f / sum;
        float o = 0.f;
        for (int r = 0; r < ns; ++r) {
            float w = __expf(s_a2[r * c + ch] - mx) * inv;
            float vv = bf2f(s_vf[r * c + ch]) + s_pos[r * c + ch];
            o += w * vv;
        }
        nf[(size_t)pt * 512 + ch] = o;
        nfb[(size_t)pt * 512 + ch] = f2bf(o);
    }
}

// ---------------------------------------------------------------------------
// Mean-pool over N: feat[P][512] -> pooled[8][512]
// ---------------------------------------------------------------------------
__global__ void pool_kernel(const float* __restrict__ feat, float* __restrict__ pooled) {
    int b = blockIdx.x;
    int ch = threadIdx.x;          // 512 threads
    float s = 0.f;
    for (int n = 0; n < 1024; ++n) s += feat[(size_t)(b * 1024 + n) * 512 + ch];
    pooled[b * 512 + ch] = s * (1.f / 1024.f);
}

// ---------------------------------------------------------------------------
// Classifier head: 512 -> 256 -> 128 -> 40, one block per batch.
// ---------------------------------------------------------------------------
__global__ __launch_bounds__(256) void cls_kernel(
    const float* __restrict__ pooled,
    const float* __restrict__ W1, const float* __restrict__ b1,
    const float* __restrict__ g1, const float* __restrict__ be1,
    const float* __restrict__ W2, const float* __restrict__ b2,
    const float* __restrict__ g2, const float* __restrict__ be2,
    const float* __restrict__ W3, const float* __restrict__ b3,
    float* __restrict__ out) {
    __shared__ float p[512];
    __shared__ float h1[256];
    __shared__ float h2[128];
    int b = blockIdx.x, t = threadIdx.x;
    p[t] = pooled[b * 512 + t];
    p[t + 256] = pooled[b * 512 + t + 256];
    __syncthreads();
    {
        float s = b1[t];
        for (int k = 0; k < 512; ++k) s += p[k] * W1[k * 256 + t];
        s = s * (g1[t] * BN_RS) + be1[t];
        h1[t] = fmaxf(s, 0.f);
    }
    __syncthreads();
    if (t < 128) {
        float s = b2[t];
        for (int k = 0; k < 256; ++k) s += h1[k] * W2[k * 128 + t];
        s = s * (g2[t] * BN_RS) + be2[t];
        h2[t] = fmaxf(s, 0.f);
    }
    __syncthreads();
    if (t < 40) {
        float s = b3[t];
        for (int k = 0; k < 128; ++k) s += h2[k] * W3[k * 40 + t];
        out[b * 40 + t] = s;
    }
}

// ---------------------------------------------------------------------------
// Host orchestration
// ---------------------------------------------------------------------------
extern "C" void kernel_launch(void* const* d_in, const int* in_sizes, int n_in,
                              void* d_out, int out_size, void* d_ws, size_t ws_size,
                              hipStream_t stream) {
    (void)in_sizes; (void)n_in; (void)out_size; (void)ws_size;
    const int planes[5] = {32, 64, 128, 256, 512};
    const int nsamp[5]  = {8, 16, 16, 16, 16};
    const int P = 8 * 1024;

    auto F = [&](int i) { return (const float*)d_in[i]; };
    const float* x = F(0);
    const float *tdW[5], *tdg[5], *tdbp[5], *Wq[5], *Wk[5], *Wv[5],
                *pW1[5], *pb1[5], *pW2[5], *pb2[5],
                *aW1[5], *ab1[5], *aW2[5], *ab2[5], *fW[5], *fbv[5];
    int ib = 1;
    for (int i = 0; i < 5; ++i) {
        tdW[i] = F(ib + 0); tdg[i] = F(ib + 1); tdbp[i] = F(ib + 2);
        Wq[i] = F(ib + 3); Wk[i] = F(ib + 4); Wv[i] = F(ib + 5);
        pW1[i] = F(ib + 6); pb1[i] = F(ib + 7); pW2[i] = F(ib + 8); pb2[i] = F(ib + 9);
        aW1[i] = F(ib + 10); ab1[i] = F(ib + 11); aW2[i] = F(ib + 12); ab2[i] = F(ib + 13);
        fW[i] = F(ib + 14); fbv[i] = F(ib + 15);
        ib += 16;
    }
    const float* cW1 = F(ib + 0); const float* cb1 = F(ib + 1);
    const float* cg1 = F(ib + 2); const float* cbe1 = F(ib + 3);
    const float* cW2 = F(ib + 4); const float* cb2 = F(ib + 5);
    const float* cg2 = F(ib + 6); const float* cbe2 = F(ib + 7);
    const float* cW3 = F(ib + 8); const float* cb3 = F(ib + 9);

    // ---- workspace carve ----
    char* ws = (char*)d_ws;
    size_t cur = 0;
    auto alloc = [&](size_t bytes) -> char* {
        cur = (cur + 255) & ~(size_t)255;
        char* p = ws + cur;
        cur += bytes;
        return p;
    };
    float* f32a = (float*)alloc((size_t)P * 512 * 4);
    float* f32b = (float*)alloc((size_t)P * 512 * 4);
    float* qf   = (float*)alloc((size_t)P * 512 * 4);
    float* nff  = (float*)alloc((size_t)P * 512 * 4);
    u16* bfa = (u16*)alloc((size_t)P * 512 * 2);
    u16* bfb = (u16*)alloc((size_t)P * 512 * 2);
    u16* nfb = (u16*)alloc((size_t)P * 512 * 2);
    u16* kfb = (u16*)alloc((size_t)P * 512 * 2);
    u16* vfb = (u16*)alloc((size_t)P * 512 * 2);
    int* idx8  = (int*)alloc((size_t)P * 16 * 4);
    int* idx16 = (int*)alloc((size_t)P * 16 * 4);
    float* xyz = (float*)alloc((size_t)P * 3 * 4);
    float* pooled = (float*)alloc(8 * 512 * 4);

    // ---- weight prep (bf16, transposed to [N][K]) ----
    u16 *tdWt[5], *WqT[5], *WkT[5], *WvT[5], *pW2T[5], *aW1T[5], *aW2T[5], *fWT[5];
    int in_c = 6;
    int Kp_td[5];
    for (int i = 0; i < 5; ++i) {
        int c = planes[i];
        int Kp = in_c < 32 ? 32 : in_c;
        Kp_td[i] = Kp;
        tdWt[i] = (u16*)alloc((size_t)c * Kp * 2);
        WqT[i]  = (u16*)alloc((size_t)c * c * 2);
        WkT[i]  = (u16*)alloc((size_t)c * c * 2);
        WvT[i]  = (u16*)alloc((size_t)c * c * 2);
        pW2T[i] = (u16*)alloc((size_t)c * 64 * 2);
        aW1T[i] = (u16*)alloc((size_t)4 * c * c * 2);
        aW2T[i] = (u16*)alloc((size_t)c * 4 * c * 2);
        fWT[i]  = (u16*)alloc((size_t)c * c * 2);
        auto tr = [&](const float* Wsrc, u16* Wdst, int K, int Nn, int KP) {
            int tot = Nn * KP;
            tr_kernel<<<dim3((tot + 255) / 256), dim3(256), 0, stream>>>(Wsrc, Wdst, K, Nn, KP);
        };
        tr(tdW[i], tdWt[i], in_c, c, Kp);
        tr(Wq[i], WqT[i], c, c, c);
        tr(Wk[i], WkT[i], c, c, c);
        tr(Wv[i], WvT[i], c, c, c);
        tr(pW2[i], pW2T[i], 64, c, 64);
        tr(aW1[i], aW1T[i], c, 4 * c, c);
        tr(aW2[i], aW2T[i], 4 * c, c, 4 * c);
        tr(fW[i], fWT[i], c, c, c);
        in_c = c;
    }

    // ---- xyz / initial features / knn ----
    init_kernel<<<dim3(P), dim3(512), 0, stream>>>(x, xyz, bfa);
    knn_kernel<8><<<dim3(32), dim3(256), 0, stream>>>(xyz, idx8);
    knn_kernel<16><<<dim3(32), dim3(256), 0, stream>>>(xyz, idx16);

    float* curF = f32a; u16* curB = bfa;
    float* nxtF = f32b; u16* nxtB = bfb;

    in_c = 6;
    for (int i = 0; i < 5; ++i) {
        int c = planes[i];
        int ns = nsamp[i];
        const int* idx = (ns == 8) ? idx8 : idx16;
        int cshift = 31 - __builtin_clz(c >> 3);

        // transition down: curB -> nxt (becomes new current feat)
        td_kernel<<<dim3(P), dim3(256), 0, stream>>>(curB, idx, tdWt[i], tdg[i], tdbp[i],
                                                     nxtF, nxtB, c, Kp_td[i], ns);
        { float* tf = curF; curF = nxtF; nxtF = tf; u16* tb = curB; curB = nxtB; nxtB = tb; }

        dim3 ggrid(c / 32, P / 256);
        // q = feat @ Wq (f32 only)
        gemm_kernel<<<ggrid, dim3(256), 0, stream>>>(curB, 512, WqT[i], c, nullptr,
                                                     nullptr, 0, qf, 512, nullptr, 0,
                                                     P, c, 0);
        // kf = feat @ Wk (bf16 only, for gather)
        gemm_kernel<<<ggrid, dim3(256), 0, stream>>>(curB, 512, WkT[i], c, nullptr,
                                                     nullptr, 0, nullptr, 0, kfb, 512,
                                                     P, c, 0);
        // vf = feat @ Wv (bf16 only)
        gemm_kernel<<<ggrid, dim3(256), 0, stream>>>(curB, 512, WvT[i], c, nullptr,
                                                     nullptr, 0, nullptr, 0, vfb, 512,
                                                     P, c, 0);

        // fused attention -> nff / nfb
        size_t shbytes = 64 + 256 + 2048
                       + (size_t)16 * c * 4      // pos f32
                       + (size_t)16 * c * 2      // a bf16 (aliased by vf)
                       + (size_t)16 * 4 * c * 2  // h bf16 (aliased by kf)
                       + (size_t)16 * c * 4;     // a2 f32
        attn_kernel<<<dim3(P), dim3(256), shbytes, stream>>>(
            idx, xyz, qf, kfb, vfb,
            pW1[i], pb1[i], pW2T[i], pb2[i],
            aW1T[i], ab1[i], aW2T[i], ab2[i],
            nff, nfb, c, ns, cshift);

        // feat' = nf @ fW + fb + feat (residual)
        gemm_kernel<<<ggrid, dim3(256), 0, stream>>>(nfb, 512, fWT[i], c, fbv[i],
                                                     curF, 512, nxtF, 512, nxtB, 512,
                                                     P, c, 0);
        { float* tf = curF; curF = nxtF; nxtF = tf; u16* tb = curB; curB = nxtB; nxtB = tb; }
        in_c = c;
    }

    // ---- head ----
    pool_kernel<<<dim3(8), dim3(512), 0, stream>>>(curF, pooled);
    cls_kernel<<<dim3(8), dim3(256), 0, stream>>>(pooled, cW1, cb1, cg1, cbe1,
                                                  cW2, cb2, cg2, cbe2, cW3, cb3,
                                                  (float*)d_out);
}
